// LocalCapsulePoolingNetwork_77309411328709
// MI455X (gfx1250) — compile-verified, matrix operand order checked by reading, not run on hardware
//
#include <hip/hip_runtime.h>
#include <hip/hip_bf16.h>
#include <math.h>

// ---------------------------------------------------------------------------
// LocalCapsulePoolingNetwork for gfx1250 (MI455X)
//   BN(->bf16) -> 3 x [ WMMA bf16 GEMM, GCN edge scatter, squash(->bf16), readout ]
//   -> MLP head -> log_softmax
// ---------------------------------------------------------------------------

typedef __attribute__((ext_vector_type(16))) __bf16 v16bf;
typedef __attribute__((ext_vector_type(8)))  __bf16 v8bf;
typedef __attribute__((ext_vector_type(8)))  float  v8f;

#define HD 128          // feature / hidden dim
#define GN 512          // number of graphs
#define BN_EPS 1e-5f

// ---- order-preserving float<->uint encoding for atomic segment-max ---------
__device__ __forceinline__ unsigned enc_f(float x) {
    unsigned u = __float_as_uint(x);
    return (u & 0x80000000u) ? ~u : (u | 0x80000000u);
}
__device__ __forceinline__ float dec_f(unsigned e) {
    return (e & 0x80000000u) ? __uint_as_float(e & 0x7FFFFFFFu)
                             : __uint_as_float(~e);
}

// ---------------------------------------------------------------------------
// 1) BatchNorm column statistics: sum / sumsq per feature (atomic partials)
// ---------------------------------------------------------------------------
__global__ void k_bn_stats(const float* __restrict__ x, float* __restrict__ csum,
                           float* __restrict__ csq, int N) {
    int f = threadIdx.x;                 // 128 threads = one per feature
    float s = 0.f, s2 = 0.f;
    for (int r = blockIdx.x; r < N; r += gridDim.x) {
        float v = x[(size_t)r * HD + f];
        s += v; s2 += v * v;
    }
    atomicAdd(&csum[f], s);
    atomicAdd(&csq[f], s2);
}

// ---------------------------------------------------------------------------
// 2) Apply BN -> bf16 activations:  hb = bf16((x-mu)*rsqrt(var+eps)*g + b)
// ---------------------------------------------------------------------------
__global__ void k_bn_apply(const float* __restrict__ x, const float* __restrict__ csum,
                           const float* __restrict__ csq, const float* __restrict__ gamma,
                           const float* __restrict__ beta, __bf16* __restrict__ hb,
                           int N) {
    size_t total = (size_t)N * HD;
    for (size_t i = blockIdx.x * (size_t)blockDim.x + threadIdx.x; i < total;
         i += (size_t)gridDim.x * blockDim.x) {
        int f = (int)(i & (HD - 1));
        float mu  = csum[f] / (float)N;
        float var = csq[f] / (float)N - mu * mu;
        hb[i] = (__bf16)((x[i] - mu) * rsqrtf(var + BN_EPS) * gamma[f] + beta[f]);
    }
}

// ---------------------------------------------------------------------------
// 3) W -> bf16, transposed to column-major:  Wt[n][k] = W[k][n]
// ---------------------------------------------------------------------------
__global__ void k_conv_wt(const float* __restrict__ W, __bf16* __restrict__ Wt) {
    int k = blockIdx.x;   // 128 rows
    int n = threadIdx.x;  // 128 cols (coalesced read)
    Wt[(size_t)n * HD + k] = (__bf16)W[(size_t)k * HD + n];
}

// ---------------------------------------------------------------------------
// 4) degree (incl. self loop), inverse-sqrt degree, per-graph counts
// ---------------------------------------------------------------------------
__global__ void k_deg(const int* __restrict__ dst, float* __restrict__ deg, int E) {
    int e = blockIdx.x * blockDim.x + threadIdx.x;
    if (e < E) atomicAdd(&deg[dst[e]], 1.0f);
}
__global__ void k_dinv(const float* __restrict__ deg, float* __restrict__ dinv, int N) {
    int i = blockIdx.x * blockDim.x + threadIdx.x;
    if (i < N) dinv[i] = rsqrtf(fmaxf(deg[i] + 1.0f, 1e-12f)); // +1 self loop
}
__global__ void k_cnt(const int* __restrict__ batch, float* __restrict__ cnt, int N) {
    int i = blockIdx.x * blockDim.x + threadIdx.x;
    if (i < N) atomicAdd(&cnt[batch[i]], 1.0f);
}

// ---------------------------------------------------------------------------
// 5) WMMA GEMM:  M[N,128] = Hb[N,128](bf16) x Wt[128,128]^T(bf16), f32 acc.
//    block = 256 threads (8 waves). Block -> 16-row strip, wave w -> 16-col tile.
//    Fragment layout (16-bit A/B): lane's elements 0..7 -> K = kb..kb+7,
//    elements 8..15 -> K = kb+16..kb+23  => two contiguous b128 loads each.
// ---------------------------------------------------------------------------
__global__ void k_gemm_wmma(const __bf16* __restrict__ A, const __bf16* __restrict__ Bt,
                            float* __restrict__ C, int N) {
    const int wave = threadIdx.x >> 5;
    const int lane = threadIdx.x & 31;
    const int row0 = blockIdx.x * 16;
    const int col0 = wave * 16;
    int mrow = row0 + (lane & 15);
    if (mrow >= N) mrow = N - 1;           // clamp loads, guard stores below
    const int ncol = col0 + (lane & 15);
    const int kb   = (lane < 16) ? 0 : 8;  // per-lane K base

    const __bf16* ap = A  + (size_t)mrow * HD + kb;
    const __bf16* bp = Bt + (size_t)ncol * HD + kb;

    v8f acc = {};
    #pragma unroll
    for (int k0 = 0; k0 < HD; k0 += 32) {
        v8bf aLo = *reinterpret_cast<const v8bf*>(ap + k0);
        v8bf aHi = *reinterpret_cast<const v8bf*>(ap + k0 + 16);
        v8bf bLo = *reinterpret_cast<const v8bf*>(bp + k0);
        v8bf bHi = *reinterpret_cast<const v8bf*>(bp + k0 + 16);
        v16bf a = __builtin_shufflevector(aLo, aHi, 0,1,2,3,4,5,6,7,8,9,10,11,12,13,14,15);
        v16bf b = __builtin_shufflevector(bLo, bHi, 0,1,2,3,4,5,6,7,8,9,10,11,12,13,14,15);
        acc = __builtin_amdgcn_wmma_f32_16x16x32_bf16(
                  false, a, false, b, (short)0, acc, false, false);
    }
    // C layout: VGPR r holds row (r + (lane<16?0:8)), col = lane&15
    const int rb = (lane < 16) ? 0 : 8;
    float* cp = C + (size_t)(row0 + rb) * HD + ncol;
    if (row0 + 16 <= N) {                  // block-uniform fast path: no guards
        #pragma unroll
        for (int r = 0; r < 8; ++r) cp[(size_t)r * HD] = acc[r];
    } else {                               // tail strip only
        #pragma unroll
        for (int r = 0; r < 8; ++r)
            if (row0 + rb + r < N) cp[(size_t)r * HD] = acc[r];
    }
}

// ---------------------------------------------------------------------------
// 6) agg init with self-loop contribution: agg[n,f] = dinv[n]^2 * m[n,f]
// ---------------------------------------------------------------------------
__global__ void k_agg_init(const float* __restrict__ m, const float* __restrict__ dinv,
                           float* __restrict__ agg, int N) {
    size_t total = (size_t)N * HD;
    for (size_t i = blockIdx.x * (size_t)blockDim.x + threadIdx.x; i < total;
         i += (size_t)gridDim.x * blockDim.x) {
        float di = dinv[i >> 7];
        agg[i] = di * di * m[i];
    }
}

// ---------------------------------------------------------------------------
// 7) edge scatter: agg[dst] += dinv[src]*dinv[dst] * m[src]
//    one wave per edge, 4 floats per lane (float4 gather, f32 atomic scatter)
// ---------------------------------------------------------------------------
__global__ void k_edge_scatter(const int* __restrict__ src, const int* __restrict__ dst,
                               const float* __restrict__ dinv, const float* __restrict__ m,
                               float* __restrict__ agg, int E) {
    int tid = blockIdx.x * blockDim.x + threadIdx.x;
    int e = tid >> 5;
    if (e >= E) return;
    int q = (tid & 31) << 2;
    int s = src[e], d = dst[e];
    float norm = dinv[s] * dinv[d];
    const float4 mv = *reinterpret_cast<const float4*>(m + (size_t)s * HD + q);
    float* ad = agg + (size_t)d * HD + q;
    atomicAdd(ad + 0, norm * mv.x);
    atomicAdd(ad + 1, norm * mv.y);
    atomicAdd(ad + 2, norm * mv.z);
    atomicAdd(ad + 3, norm * mv.w);
}

// ---------------------------------------------------------------------------
// 8) fused: agg+b -> squash -> hb(bf16) ; attn = h . tar_w ; 3-way readout
//    one wave per node (4 features / lane), shuffle reductions
// ---------------------------------------------------------------------------
__global__ void k_squash_readout(const float* __restrict__ agg, const float* __restrict__ bias,
                                 const float* __restrict__ tarw, const int* __restrict__ batch,
                                 __bf16* __restrict__ hb, float* __restrict__ wsum,
                                 float* __restrict__ sumb, unsigned* __restrict__ maxb,
                                 int N) {
    int node = blockIdx.x * (blockDim.x >> 5) + (threadIdx.x >> 5);
    if (node >= N) return;
    int lane = threadIdx.x & 31;
    float v[4];
    float n2 = 0.f;
    #pragma unroll
    for (int j = 0; j < 4; ++j) {
        int f = lane * 4 + j;
        float a = agg[(size_t)node * HD + f] + bias[f];
        v[j] = a;
        n2 += a * a;
    }
    #pragma unroll
    for (int off = 16; off > 0; off >>= 1) n2 += __shfl_xor(n2, off, 32);
    float scale = (n2 / (1.0f + n2)) * rsqrtf(n2 + 1e-12f);

    float attn = 0.f;
    #pragma unroll
    for (int j = 0; j < 4; ++j) {
        int f = lane * 4 + j;
        v[j] *= scale;
        hb[(size_t)node * HD + f] = (__bf16)v[j];
        attn += v[j] * tarw[f];
    }
    #pragma unroll
    for (int off = 16; off > 0; off >>= 1) attn += __shfl_xor(attn, off, 32);

    int g = batch[node];
    #pragma unroll
    for (int j = 0; j < 4; ++j) {
        int f = lane * 4 + j;
        atomicAdd(&wsum[(size_t)g * HD + f], attn * v[j]);
        atomicAdd(&sumb[(size_t)g * HD + f], v[j]);
        atomicMax(&maxb[(size_t)g * HD + f], enc_f(v[j]));
    }
}

// ---------------------------------------------------------------------------
// 9) per-layer readout accumulation into graph_rep[G, 384]
// ---------------------------------------------------------------------------
__global__ void k_init_max(unsigned* __restrict__ maxb, int n) {
    int i = blockIdx.x * blockDim.x + threadIdx.x;
    if (i < n) maxb[i] = 0x007FFFFFu;   // enc_f(-inf)
}
__global__ void k_readout_accum(const float* __restrict__ wsum, const float* __restrict__ sumb,
                                const unsigned* __restrict__ maxb, const float* __restrict__ cnt,
                                float* __restrict__ gr) {
    int g = blockIdx.x, f = threadIdx.x;
    float c = cnt[g];
    float cs = fmaxf(c, 1.0f);
    gr[(size_t)g * 384 + f]       += wsum[(size_t)g * HD + f];
    gr[(size_t)g * 384 + 128 + f] += sumb[(size_t)g * HD + f] / cs;
    float mx = dec_f(maxb[(size_t)g * HD + f]);
    gr[(size_t)g * 384 + 256 + f] += (c > 0.f) ? mx : 0.f;
}

// ---------------------------------------------------------------------------
// 10) MLP head + log_softmax (C=2). One block per graph.
// ---------------------------------------------------------------------------
__global__ void k_head(const float* __restrict__ gr, const float* __restrict__ W1,
                       const float* __restrict__ b1, const float* __restrict__ W2,
                       const float* __restrict__ b2, float* __restrict__ out) {
    __shared__ float z1[HD];
    __shared__ float red[HD];
    __shared__ float zz[2];
    int g = blockIdx.x, t = threadIdx.x;
    float acc = b1[t];
    const float* gv = gr + (size_t)g * 384;
    for (int k = 0; k < 384; ++k) acc += gv[k] * W1[(size_t)k * HD + t];
    z1[t] = fmaxf(acc, 0.f);
    __syncthreads();
    for (int c = 0; c < 2; ++c) {
        red[t] = z1[t] * W2[t * 2 + c];
        __syncthreads();
        for (int s = 64; s > 0; s >>= 1) {
            if (t < s) red[t] += red[t + s];
            __syncthreads();
        }
        if (t == 0) zz[c] = red[0] + b2[c];
        __syncthreads();
    }
    if (t < 2) {
        float mz = fmaxf(zz[0], zz[1]);
        float lse = mz + logf(expf(zz[0] - mz) + expf(zz[1] - mz));
        out[(size_t)g * 2 + t] = zz[t] - lse;
    }
}

// ---------------------------------------------------------------------------
extern "C" void kernel_launch(void* const* d_in, const int* in_sizes, int n_in,
                              void* d_out, int out_size, void* d_ws, size_t ws_size,
                              hipStream_t stream) {
    const float* x      = (const float*)d_in[0];
    const int*   ei     = (const int*)  d_in[1];
    const int*   batch  = (const int*)  d_in[2];
    const float* gamma  = (const float*)d_in[3];
    const float* beta   = (const float*)d_in[4];
    const float* Wl[3]  = {(const float*)d_in[5], (const float*)d_in[7], (const float*)d_in[9]};
    const float* bl[3]  = {(const float*)d_in[6], (const float*)d_in[8], (const float*)d_in[10]};
    const float* tarw   = (const float*)d_in[11];
    const float* lin1_w = (const float*)d_in[12];
    const float* lin1_b = (const float*)d_in[13];
    const float* lin2_w = (const float*)d_in[14];
    const float* lin2_b = (const float*)d_in[15];
    float* out = (float*)d_out;

    const int N = in_sizes[0] / HD;
    const int E = in_sizes[1] / 2;
    const int* src = ei;
    const int* dst = ei + E;

    // ---- workspace layout (float-granular offsets, all 16B aligned) ----
    float* W = (float*)d_ws;
    size_t o = 0;
    float*    m     = W + o; o += (size_t)N * HD;        // GEMM output (f32)
    float*    agg   = W + o; o += (size_t)N * HD;        // scatter accumulator (f32)
    __bf16*   hb    = (__bf16*)(W + o); o += (size_t)N * HD / 2;  // activations (bf16)
    __bf16*   Wt    = (__bf16*)(W + o); o += (size_t)HD * HD / 2; // W^T (bf16)
    float*    deg   = W + o; o += (size_t)N;
    float*    dinv  = W + o; o += (size_t)N;
    float*    csum  = W + o; o += HD;
    float*    csq   = W + o; o += HD;
    float*    cnt   = W + o; o += GN;
    float*    wsum  = W + o; o += (size_t)GN * HD;
    float*    sumb  = W + o; o += (size_t)GN * HD;
    unsigned* maxb  = (unsigned*)(W + o); o += (size_t)GN * HD;
    float*    gr    = W + o; o += (size_t)GN * 384;

    // ---- zero-init (graph-capturable memset nodes) ----
    hipMemsetAsync(csum, 0, 2 * HD * sizeof(float), stream);   // csum+csq contiguous
    hipMemsetAsync(deg,  0, (size_t)N * sizeof(float), stream);
    hipMemsetAsync(cnt,  0, GN * sizeof(float), stream);
    hipMemsetAsync(gr,   0, (size_t)GN * 384 * sizeof(float), stream);

    // ---- BatchNorm -> bf16 activations ----
    k_bn_stats<<<1024, HD, 0, stream>>>(x, csum, csq, N);
    k_bn_apply<<<4096, 256, 0, stream>>>(x, csum, csq, gamma, beta, hb, N);

    // ---- GCN normalization data ----
    k_deg <<<(E + 255) / 256, 256, 0, stream>>>(dst, deg, E);
    k_dinv<<<(N + 255) / 256, 256, 0, stream>>>(deg, dinv, N);
    k_cnt <<<(N + 255) / 256, 256, 0, stream>>>(batch, cnt, N);

    const int gemm_blocks    = (N + 15) / 16;
    const int scatter_blocks = ((E * 32) + 255) / 256;
    const int squash_blocks  = (N + 7) / 8;

    for (int layer = 0; layer < 3; ++layer) {
        hipMemsetAsync(wsum, 0, 2 * (size_t)GN * HD * sizeof(float), stream); // wsum+sumb
        k_init_max<<<(GN * HD + 255) / 256, 256, 0, stream>>>(maxb, GN * HD);

        k_conv_wt  <<<HD, HD, 0, stream>>>(Wl[layer], Wt);
        k_gemm_wmma<<<gemm_blocks, 256, 0, stream>>>(hb, Wt, m, N);
        k_agg_init <<<4096, 256, 0, stream>>>(m, dinv, agg, N);
        k_edge_scatter<<<scatter_blocks, 256, 0, stream>>>(src, dst, dinv, m, agg, E);
        k_squash_readout<<<squash_blocks, 256, 0, stream>>>(
            agg, bl[layer], tarw + layer * HD, batch, hb, wsum, sumb, maxb, N);
        k_readout_accum<<<GN, HD, 0, stream>>>(wsum, sumb, maxb, cnt, gr);
    }

    k_head<<<GN, HD, 0, stream>>>(gr, lin1_w, lin1_b, lin2_w, lin2_b, out);
    (void)n_in; (void)out_size; (void)ws_size;
}
